// Grok1SparseMoeBlock_17136919511387
// MI455X (gfx1250) — matrix operand order repaired and checked by reading, not compiled, not run on hardware
//
#include <hip/hip_runtime.h>
#include <math.h>

// ---------------------------------------------------------------------------
// Grok-1 sparse MoE block for MI455X (gfx1250, wave32, WMMA bf16).
// final = sum_{k in top2} softmax_w_k * ( gelu(x@w0_e) * (x@wv_e) ) @ w1_e
// router_logits = x @ gate_w   (fp32 exact)
// ---------------------------------------------------------------------------

#define S_  2048
#define H_  1024
#define I_  4096
#define E_  8
#define MAXROWS 4352   // 2*S + E*32 padding slack

typedef __bf16 v16bf __attribute__((ext_vector_type(16)));
typedef __bf16 v8bf  __attribute__((ext_vector_type(8)));
typedef float  v8f   __attribute__((ext_vector_type(8)));

__device__ __forceinline__ unsigned bfbits(float f) {
    unsigned u = __builtin_bit_cast(unsigned, f);
    return u + 0x7FFFu + ((u >> 16) & 1u);            // round-to-nearest-even
}
// pack two fp32 -> packed bf16 pair via V_PERM_B32 (single merge instruction)
__device__ __forceinline__ unsigned pack2bf(float lo, float hi) {
    return __builtin_amdgcn_perm(bfbits(hi), bfbits(lo), 0x07060302u);
}
__device__ __forceinline__ __bf16 f2bf(float f) {
    unsigned short h = (unsigned short)(bfbits(f) >> 16);
    return __builtin_bit_cast(__bf16, h);
}
__device__ __forceinline__ float gelu_tanh(float x) {
    const float c = 0.7978845608028654f;               // sqrt(2/pi)
    return 0.5f * x * (1.0f + tanhf(c * (x + 0.044715f * x * x * x)));
}
// A fragment: two contiguous b128 LDS reads + compile-time shuffle
__device__ __forceinline__ v16bf load_a_frag(const __bf16* rowp, int kb) {
    v8bf lo = *(const v8bf*)(rowp + kb);
    v8bf hi = *(const v8bf*)(rowp + 8 + kb);
    return __builtin_shufflevector(lo, hi, 0, 1, 2, 3, 4, 5, 6, 7,
                                           8, 9, 10, 11, 12, 13, 14, 15);
}

// ---------------------------------------------------------------------------
// Workspace layout (bytes)
// ---------------------------------------------------------------------------
constexpr size_t OFF_TOPI = 0;                         // int  [S*2]
constexpr size_t OFF_TOPW = OFF_TOPI + S_ * 2 * 4;     // f32  [S*2]
constexpr size_t OFF_CNT  = OFF_TOPW + S_ * 2 * 4;     // int  [E]
constexpr size_t OFF_PCNT = OFF_CNT + 256;             // int  [E]
constexpr size_t OFF_BASE = OFF_PCNT + 256;            // int  [E]
constexpr size_t OFF_TOK  = OFF_BASE + 256;            // int  [MAXROWS]
constexpr size_t OFF_WGT  = OFF_TOK + MAXROWS * 4;     // f32  [MAXROWS]
constexpr size_t OFF_SLOT = OFF_WGT + MAXROWS * 4;     // int  [S*2]
constexpr size_t OFF_ACT  = (OFF_SLOT + S_ * 2 * 4 + 255) & ~size_t(255); // bf16 [MAXROWS*I]
constexpr size_t OFF_Y    = (OFF_ACT + (size_t)MAXROWS * I_ * 2 + 255) & ~size_t(255); // f32 [MAXROWS*H]

// ---------------------------------------------------------------------------
// 1) Router: one wave32 per token.  logits -> d_out tail, top-2 softmax -> ws.
// ---------------------------------------------------------------------------
__global__ __launch_bounds__(256)
void router_kernel(const float* __restrict__ X, const float* __restrict__ GW,
                   float* __restrict__ logits_out,
                   int* __restrict__ top_idx, float* __restrict__ top_w) {
    const int wave = threadIdx.x >> 5;
    const int lane = threadIdx.x & 31;
    const int t = blockIdx.x * 8 + wave;
    const float* x = X + (size_t)t * H_;

    float acc[E_];
#pragma unroll
    for (int e = 0; e < E_; ++e) acc[e] = 0.0f;

    for (int h = lane; h < H_; h += 32) {
        const float xv = x[h];
        const float* g = GW + (size_t)h * E_;
#pragma unroll
        for (int e = 0; e < E_; ++e) acc[e] += xv * g[e];
    }
#pragma unroll
    for (int off = 16; off >= 1; off >>= 1) {
#pragma unroll
        for (int e = 0; e < E_; ++e) acc[e] += __shfl_xor(acc[e], off, 32);
    }
    if (lane == 0) {
#pragma unroll
        for (int e = 0; e < E_; ++e) logits_out[(size_t)t * E_ + e] = acc[e];
        int i1 = 0; float m1 = acc[0];
#pragma unroll
        for (int e = 1; e < E_; ++e) if (acc[e] > m1) { m1 = acc[e]; i1 = e; }
        int i2 = (i1 == 0) ? 1 : 0; float m2 = acc[i2];
#pragma unroll
        for (int e = 0; e < E_; ++e)
            if (e != i1 && acc[e] > m2) { m2 = acc[e]; i2 = e; }
        const float e2 = expf(m2 - m1);
        const float inv = 1.0f / (1.0f + e2);
        top_idx[t * 2 + 0] = i1;  top_w[t * 2 + 0] = inv;
        top_idx[t * 2 + 1] = i2;  top_w[t * 2 + 1] = e2 * inv;
    }
}

// ---------------------------------------------------------------------------
// 2) Deterministic per-expert compaction (single block).  Pads to 32 rows.
// ---------------------------------------------------------------------------
__global__ void build_lists_kernel(const int* __restrict__ top_idx,
                                   const float* __restrict__ top_w,
                                   int* __restrict__ counts, int* __restrict__ pcounts,
                                   int* __restrict__ bases, int* __restrict__ tok_list,
                                   float* __restrict__ wgt_list, int* __restrict__ slot_of) {
    const int tid = threadIdx.x;
    if (tid < E_) {
        int c = 0;
        for (int i = 0; i < S_ * 2; ++i) c += (top_idx[i] == tid);
        counts[tid] = c;
        pcounts[tid] = (c + 31) & ~31;
    }
    __syncthreads();
    if (tid == 0) {
        int b = 0;
        for (int e = 0; e < E_; ++e) { bases[e] = b; b += pcounts[e]; }
    }
    __syncthreads();
    if (tid < E_) {
        int idx = bases[tid];
        for (int i = 0; i < S_ * 2; ++i) {
            if (top_idx[i] == tid) {
                tok_list[idx] = i >> 1;
                wgt_list[idx] = top_w[i];
                slot_of[i] = idx;
                ++idx;
            }
        }
        for (int p = counts[tid]; p < pcounts[tid]; ++p) {   // zero-weight padding
            tok_list[bases[tid] + p] = 0;
            wgt_list[bases[tid] + p] = 0.0f;
        }
    }
}

// ---------------------------------------------------------------------------
// 3) GEMM1: act = gelu(Xg @ w0_e) * (Xg @ wv_e) * routing_weight   (bf16 WMMA)
//    block = 256 thr = 8 waves; tile = 32 rows x 128 cols; K-chunks of 64
//    (two WMMA K-steps per staged tile -> 8 wmma per iteration).
// ---------------------------------------------------------------------------
__global__ __launch_bounds__(256)
void moe_gemm1_kernel(const float* __restrict__ X, const float* __restrict__ W0,
                      const float* __restrict__ WV,
                      const int* __restrict__ tok_list, const float* __restrict__ wgt_list,
                      const int* __restrict__ pcounts, const int* __restrict__ bases,
                      __bf16* __restrict__ act) {
    const int e     = blockIdx.z;
    const int mtile = blockIdx.y;
    const int n0    = blockIdx.x * 128;
    if (mtile * 32 >= pcounts[e]) return;
    const int rowbase = bases[e] + mtile * 32;

    __shared__ unsigned lX[32 * 32];      // 32 rows x 64 bf16 (packed pairs)
    __shared__ unsigned lW0[128 * 32];    // transposed [n][k/2], 64 K per row
    __shared__ unsigned lWV[128 * 32];
    __shared__ float    wrow[32];

    const int tid = threadIdx.x, lane = tid & 31, wave = tid >> 5;
    if (tid < 32) wrow[tid] = wgt_list[rowbase + tid];

    // --- hoisted per-thread staging geometry ---
    const int sm = tid >> 3;              // X staging row 0..31
    const int sk = (tid & 7) * 8;         // X staging k   0,8,..,56
    const float* xRow = X + (size_t)tok_list[rowbase + sm] * H_;   // gathered row

    const int kp0 = tid >> 5;             // weight k-pair 0..7 (items add 8*it)
    const int n4  = (tid & 31) * 4;       // weight n group
    const size_t wb = (size_t)e * H_ * I_ + (size_t)(2 * kp0) * I_ + n0 + n4;
    const int ldsW = n4 * 32 + kp0;       // LDS dword index, items add 8*it

    v8f ch0 = {}, ch1 = {}, cv0 = {}, cv1 = {};
    __syncthreads();

    for (int k0 = 0; k0 < H_; k0 += 64) {
        // ---- stage gathered X chunk (32 x 64) as packed bf16, one b128 store ----
        {
            const float4 x0 = *(const float4*)(xRow + k0 + sk);
            const float4 x1 = *(const float4*)(xRow + k0 + sk + 4);
            uint4 pk;
            pk.x = pack2bf(x0.x, x0.y);  pk.y = pack2bf(x0.z, x0.w);
            pk.z = pack2bf(x1.x, x1.y);  pk.w = pack2bf(x1.z, x1.w);
            *(uint4*)&lX[sm * 32 + (sk >> 1)] = pk;
        }
        // ---- stage W0/WV chunks (64k x 128n) K-transposed; items differ by
        //      constant offsets so loads fold into instruction IOFFSETs ----
        const float* p0 = W0 + wb + (size_t)k0 * I_;
        const float* pv = WV + wb + (size_t)k0 * I_;
#pragma unroll
        for (int it = 0; it < 4; ++it) {
            const size_t off = (size_t)(16 * it) * I_;
            const int    li  = ldsW + 8 * it;
            const float4 a0 = *(const float4*)(p0 + off);
            const float4 a1 = *(const float4*)(p0 + off + I_);
            lW0[li +  0] = pack2bf(a0.x, a1.x);
            lW0[li + 32] = pack2bf(a0.y, a1.y);
            lW0[li + 64] = pack2bf(a0.z, a1.z);
            lW0[li + 96] = pack2bf(a0.w, a1.w);
            const float4 v0 = *(const float4*)(pv + off);
            const float4 v1 = *(const float4*)(pv + off + I_);
            lWV[li +  0] = pack2bf(v0.x, v1.x);
            lWV[li + 32] = pack2bf(v0.y, v1.y);
            lWV[li + 64] = pack2bf(v0.z, v1.z);
            lWV[li + 96] = pack2bf(v0.w, v1.w);
        }
        if (k0 + 64 < H_) {   // stream next weight chunk toward L2 (global_prefetch_b8)
            __builtin_prefetch(p0 + (size_t)64 * I_, 0, 1);
            __builtin_prefetch(pv + (size_t)64 * I_, 0, 1);
        }
        __syncthreads();

        const __bf16* pA0 = (const __bf16*)&lX[(lane & 15) * 32];
        const __bf16* pA1 = (const __bf16*)&lX[((lane & 15) + 16) * 32];
        const __bf16* pB0 = (const __bf16*)&lW0[(wave * 16 + (lane & 15)) * 32];
        const __bf16* pBv = (const __bf16*)&lWV[(wave * 16 + (lane & 15)) * 32];
        const int kbA = (lane < 16) ? 0 : 16;
        const int kbB = (lane < 16) ? 0 : 16;
#pragma unroll
        for (int s = 0; s < 2; ++s) {           // two WMMA K-steps per staged tile
            const v16bf a0 = load_a_frag(pA0 + s * 32, kbA);
            const v16bf a1 = load_a_frag(pA1 + s * 32, kbA);
            const v16bf b0 = *(const v16bf*)(pB0 + s * 32 + kbB);
            const v16bf bv = *(const v16bf*)(pBv + s * 32 + kbB);
            ch0 = __builtin_amdgcn_wmma_f32_16x16x32_bf16(false, a0, false, b0, (short)0, ch0, false, false);
            ch1 = __builtin_amdgcn_wmma_f32_16x16x32_bf16(false, a1, false, b0, (short)0, ch1, false, false);
            cv0 = __builtin_amdgcn_wmma_f32_16x16x32_bf16(false, a0, false, bv, (short)0, cv0, false, false);
            cv1 = __builtin_amdgcn_wmma_f32_16x16x32_bf16(false, a1, false, bv, (short)0, cv1, false, false);
        }
        __syncthreads();
    }

    // ---- epilogue: fused gelu(h)*v*weight, compacted bf16 activation rows ----
    const int nn = n0 + wave * 16 + (lane & 15);
    const int mh = (lane >> 4) << 3;
#pragma unroll
    for (int r = 0; r < 8; ++r) {
        const int m0 = mh + r;
        const int m1 = mh + r + 16;
        act[(size_t)(rowbase + m0) * I_ + nn] = f2bf(gelu_tanh(ch0[r]) * cv0[r] * wrow[m0]);
        act[(size_t)(rowbase + m1) * I_ + nn] = f2bf(gelu_tanh(ch1[r]) * cv1[r] * wrow[m1]);
    }
}

// ---------------------------------------------------------------------------
// 4) GEMM2: y = act(bf16) @ w1_e(->bf16), fp32 out to ybuf.  K-chunks of 64.
// ---------------------------------------------------------------------------
__global__ __launch_bounds__(256)
void moe_gemm2_kernel(const __bf16* __restrict__ act, const float* __restrict__ W1,
                      const int* __restrict__ pcounts, const int* __restrict__ bases,
                      float* __restrict__ ybuf) {
    const int e     = blockIdx.z;
    const int mtile = blockIdx.y;
    const int n0    = blockIdx.x * 128;
    if (mtile * 32 >= pcounts[e]) return;
    const int rowbase = bases[e] + mtile * 32;

    __shared__ unsigned lA[32 * 32];      // 32 rows x 64 bf16
    __shared__ unsigned lW[128 * 32];     // transposed [n][k/2]

    const int tid = threadIdx.x, lane = tid & 31, wave = tid >> 5;

    const int sm = tid >> 3;
    const int sk = (tid & 7) * 8;
    const __bf16* aRow = act + (size_t)(rowbase + sm) * I_;

    const int kp0 = tid >> 5;
    const int n4  = (tid & 31) * 4;
    const size_t wb = (size_t)e * I_ * H_ + (size_t)(2 * kp0) * H_ + n0 + n4;
    const int ldsW = n4 * 32 + kp0;

    v8f c0 = {}, c1 = {};

    for (int k0 = 0; k0 < I_; k0 += 64) {
        // ---- stage act chunk (already bf16): one b128 load + b128 store ----
        {
            const uint4 v = *(const uint4*)(aRow + k0 + sk);
            *(uint4*)&lA[sm * 32 + (sk >> 1)] = v;
        }
        // ---- stage W1 chunk (64k x 128n) K-transposed ----
        const float* p1 = W1 + wb + (size_t)k0 * H_;
#pragma unroll
        for (int it = 0; it < 4; ++it) {
            const size_t off = (size_t)(16 * it) * H_;
            const int    li  = ldsW + 8 * it;
            const float4 a0 = *(const float4*)(p1 + off);
            const float4 a1 = *(const float4*)(p1 + off + H_);
            lW[li +  0] = pack2bf(a0.x, a1.x);
            lW[li + 32] = pack2bf(a0.y, a1.y);
            lW[li + 64] = pack2bf(a0.z, a1.z);
            lW[li + 96] = pack2bf(a0.w, a1.w);
        }
        if (k0 + 64 < I_)
            __builtin_prefetch(p1 + (size_t)64 * H_, 0, 1);
        __syncthreads();

        const __bf16* pA0 = (const __bf16*)&lA[(lane & 15) * 32];
        const __bf16* pA1 = (const __bf16*)&lA[((lane & 15) + 16) * 32];
        const __bf16* pB  = (const __bf16*)&lW[(wave * 16 + (lane & 15)) * 32];
        const int kbA = (lane < 16) ? 0 : 16;
        const int kbB = (lane < 16) ? 0 : 16;
#pragma unroll
        for (int s = 0; s < 2; ++s) {
            const v16bf a0 = load_a_frag(pA0 + s * 32, kbA);
            const v16bf a1 = load_a_frag(pA1 + s * 32, kbA);
            const v16bf b  = *(const v16bf*)(pB + s * 32 + kbB);
            c0 = __builtin_amdgcn_wmma_f32_16x16x32_bf16(false, a0, false, b, (short)0, c0, false, false);
            c1 = __builtin_amdgcn_wmma_f32_16x16x32_bf16(false, a1, false, b, (short)0, c1, false, false);
        }
        __syncthreads();
    }

    const int nn = n0 + wave * 16 + (lane & 15);
    const int mh = (lane >> 4) << 3;
#pragma unroll
    for (int r = 0; r < 8; ++r) {
        ybuf[(size_t)(rowbase + mh + r) * H_ + nn]      = c0[r];
        ybuf[(size_t)(rowbase + mh + r + 16) * H_ + nn] = c1[r];
    }
}

// ---------------------------------------------------------------------------
// 5) Combine: final[t] = y[slot(t,0)] + y[slot(t,1)]  (weights already folded)
// ---------------------------------------------------------------------------
__global__ __launch_bounds__(256)
void combine_kernel(const float* __restrict__ ybuf, const int* __restrict__ slot_of,
                    float* __restrict__ out) {
    const size_t h4 = ((size_t)blockIdx.x * 256 + threadIdx.x) * 4;   // S*H/4 threads
    const int t = (int)(h4 >> 10);
    const int h = (int)(h4 & 1023);
    const int s0 = slot_of[t * 2 + 0];
    const int s1 = slot_of[t * 2 + 1];
    const float4 y0 = *(const float4*)(ybuf + (size_t)s0 * H_ + h);
    const float4 y1 = *(const float4*)(ybuf + (size_t)s1 * H_ + h);
    float4 r;
    r.x = y0.x + y1.x; r.y = y0.y + y1.y; r.z = y0.z + y1.z; r.w = y0.w + y1.w;
    *(float4*)(out + (size_t)t * H_ + h) = r;
}

// ---------------------------------------------------------------------------
extern "C" void kernel_launch(void* const* d_in, const int* in_sizes, int n_in,
                              void* d_out, int out_size, void* d_ws, size_t ws_size,
                              hipStream_t stream) {
    const float* X  = (const float*)d_in[0];   // [1,2048,1024]
    const float* GW = (const float*)d_in[1];   // [1024,8]
    const float* W0 = (const float*)d_in[2];   // [8,1024,4096]
    const float* WV = (const float*)d_in[3];   // [8,1024,4096]
    const float* W1 = (const float*)d_in[4];   // [8,4096,1024]
    float* out = (float*)d_out;
    char*  ws  = (char*)d_ws;

    int*    top_idx  = (int*)   (ws + OFF_TOPI);
    float*  top_w    = (float*) (ws + OFF_TOPW);
    int*    counts   = (int*)   (ws + OFF_CNT);
    int*    pcounts  = (int*)   (ws + OFF_PCNT);
    int*    bases    = (int*)   (ws + OFF_BASE);
    int*    tok_list = (int*)   (ws + OFF_TOK);
    float*  wgt_list = (float*) (ws + OFF_WGT);
    int*    slot_of  = (int*)   (ws + OFF_SLOT);
    __bf16* act      = (__bf16*)(ws + OFF_ACT);
    float*  ybuf     = (float*) (ws + OFF_Y);

    float* logits_out = out + (size_t)S_ * H_;   // router_logits after `final`

    router_kernel<<<S_ / 8, 256, 0, stream>>>(X, GW, logits_out, top_idx, top_w);
    build_lists_kernel<<<1, 64, 0, stream>>>(top_idx, top_w, counts, pcounts, bases,
                                             tok_list, wgt_list, slot_of);
    {
        dim3 grid(I_ / 128, S_ / 32, E_);
        moe_gemm1_kernel<<<grid, 256, 0, stream>>>(X, W0, WV, tok_list, wgt_list,
                                                   pcounts, bases, act);
    }
    {
        dim3 grid(H_ / 128, S_ / 32, E_);
        moe_gemm2_kernel<<<grid, 256, 0, stream>>>(act, W1, pcounts, bases, ybuf);
    }
    combine_kernel<<<(S_ * H_) / 1024, 256, 0, stream>>>(ybuf, slot_of, out);
}
